// MultiViewNetwork_63307817943441
// MI455X (gfx1250) — compile-verified, matrix operand order checked by reading, not run on hardware
//
#include <hip/hip_runtime.h>
#include <hip/hip_bf16.h>
#include <math.h>
#include <stdint.h>

// ---------------------------------------------------------------------------
// MI455X / gfx1250 implementation of the multi-view network forward pass.
// Convs run as implicit-GEMM on v_wmma_f32_16x16x32_bf16 (wave32). Each wave
// computes a 32(out-ch) x 64(pixel) register tile: 2 A-frags x 4 B-frags ->
// 8 accumulators, 8 WMMAs per 32-wide K chunk. Weight tiles are DMA'd into
// LDS by the Tensor Data Mover, double-buffered (issue chunk k+1, wait
// TENSORcnt<=1, consume chunk k) and read back via ds_load_b128. All cin are
// multiples of 32 so K = cin*k*k is a multiple of 32 (no K guards).
// Gather/scatter ops are bandwidth-bound scalar kernels.
// ---------------------------------------------------------------------------

typedef __attribute__((ext_vector_type(16))) __bf16 bf16x16;
typedef __attribute__((ext_vector_type(8)))  float  f32x8;
typedef unsigned int u32;
typedef unsigned long long u64;
typedef __attribute__((ext_vector_type(4))) u32 u32x4;
typedef __attribute__((ext_vector_type(8))) int i32x8;
typedef __attribute__((ext_vector_type(4))) int i32x4;

__device__ __forceinline__ __bf16 f2bf(float f) {
  unsigned u = __float_as_uint(f);
  u += 0x7FFFu + ((u >> 16) & 1u);          // round-to-nearest-even
  unsigned short h = (unsigned short)(u >> 16);
  return __builtin_bit_cast(__bf16, h);
}

// ---------------------------------------------------------------------------
// TDM: DMA a 2D fp32 tile (tile_w x tile_h elements, row stride in elements)
// from global memory into LDS. D# packing per CDNA5 ISA 08_async_tensor §8:
//   group0: [1:0]=count=1, [63:32]=lds byte addr, [120:64]=global byte addr,
//           [127:126]=type=2
//   group1: [17:16]=data_size(2 -> 4B), [79:48]=tensor_dim0, [111:80]=dim1,
//           [127:112]=tile_dim0, [143:128]=tile_dim1, [207:160]=dim0_stride
// tensor dims == tile dims (tile fully interior, no OOB). Groups 2/3 unused.
// This toolchain exposes the 6-arg builtin:
//   (u32x4 g0, i32x8 g1, i32x4, i32x4, i32x8, i32 cpol)
// ---------------------------------------------------------------------------
__device__ __forceinline__ void tdm_load_tile_f32(u32 lds_byte_off,
                                                  const float* gsrc,
                                                  u32 tile_w, u32 tile_h,
                                                  u32 row_stride_elems) {
  const u64 ga = (u64)(uintptr_t)gsrc;
  u32x4 g0;
  g0[0] = 1u;                                    // count=1 (valid user D#)
  g0[1] = lds_byte_off;                          // lds_addr
  g0[2] = (u32)ga;                               // global_addr[31:0]
  g0[3] = (u32)(ga >> 32) | (2u << 30);          // global_addr[56:32] | type=2
  i32x8 g1;
  g1[0] = (int)(2u << 16);                       // data_size = 4 bytes
  g1[1] = (int)(tile_w << 16);                   // tensor_dim0[15:0]
  g1[2] = (int)((tile_w >> 16) | (tile_h << 16));// dim0 hi | tensor_dim1 lo
  g1[3] = (int)((tile_h >> 16) | (tile_w << 16));// dim1 hi | tile_dim0
  g1[4] = (int)tile_h;                           // tile_dim1 | tile_dim2=0
  g1[5] = (int)row_stride_elems;                 // tensor_dim0_stride[31:0]
  g1[6] = 0;                                     // stride hi | dim1_stride lo
  g1[7] = 0;
  const i32x4 z4 = {0, 0, 0, 0};
  const i32x8 z8 = {0, 0, 0, 0, 0, 0, 0, 0};
  __builtin_amdgcn_tensor_load_to_lds(g0, g1, z4, z4, z8, 0);
}

// ---------------------------------------------------------------------------
// Implicit-GEMM conv (KS = 1 or 3), optional fused residual-add + ReLU.
// Weights OIHW (contiguous in K = ci*KS*KS + ky*KS + kx).
// Fragment layouts per CDNA5 ISA 7.12.2:
//   A/B 16-bit frag, lane L: row/col = L%16, element i (0..15) holds
//     K = kc + (i<8 ? 8*(L>=16)+i : 16+8*(L>=16)+(i-8))
//   C/D: VGPR r -> (M = r + 8*(L>=16), N = L%16)
// Requires: cout % 32 == 0, (outH*outW) % 64 == 0, cin % 32 == 0.
// ---------------------------------------------------------------------------
template <int KS>
__global__ __launch_bounds__(256) void conv_wmma_k(
    const float* __restrict__ in, const float* __restrict__ wt,
    const float* __restrict__ res, float* __restrict__ out,
    int cin, int cout, int inH, int inW, int outH, int outW,
    int stride, int pad, int do_relu) {
  __shared__ float lds_w[8][2][32 * 32];             // 2 x 4KB per wave (dbuf)
  const int HW      = outH * outW;
  const int tiles_n = HW >> 6;                       // 64 pixels per wave
  const int total   = (cout >> 5) * tiles_n;         // 32 out-ch per wave
  const int wave    = blockIdx.x * (blockDim.x >> 5) + (threadIdx.x >> 5);
  if (wave >= total) return;                         // wave-uniform
  const int wid   = threadIdx.x >> 5;
  const int lane  = threadIdx.x & 31;
  const int half  = lane >> 4;
  const int half8 = half << 3;
  const int row   = lane & 15;                       // M for A, N for B
  const int tm    = wave / tiles_n;
  const int tn    = wave - tm * tiles_n;
  const int co0   = tm << 5;
  const int p0    = tn << 6;
  const int K     = cin * KS * KS;                   // multiple of 32

  // pixel coordinates for the 4 N-subtiles of this lane
  int oy[4], ox[4];
#pragma unroll
  for (int j = 0; j < 4; ++j) {
    const int p = p0 + (j << 4) + row;
    oy[j] = p / outW;
    ox[j] = p - oy[j] * outW;
  }

  const float* __restrict__ wtile = wt + (size_t)co0 * K;   // 32 rows x K
  const u32 lds_b0 = (u32)(uintptr_t)&lds_w[wid][0][0];
  const u32 lds_b1 = (u32)(uintptr_t)&lds_w[wid][1][0];

  f32x8 acc[8];
#pragma unroll
  for (int t = 0; t < 8; ++t) acc[t] = (f32x8){0.f,0.f,0.f,0.f,0.f,0.f,0.f,0.f};

  // ---- prologue: DMA first weight tile, then pipeline (double-buffered) ---
  tdm_load_tile_f32(lds_b0, wtile, 32u, 32u, (u32)K);

  int ping = 0;
  for (int kc = 0; kc < K; kc += 32) {
    if (kc + 32 < K) {       // prefetch next chunk into the other buffer
      tdm_load_tile_f32(ping ? lds_b0 : lds_b1, wtile + kc + 32, 32u, 32u,
                        (u32)K);
      __builtin_amdgcn_s_wait_tensorcnt(1);   // wait for the older transfer
    } else {
      __builtin_amdgcn_s_wait_tensorcnt(0);   // last chunk: drain all
    }
    const float* lb = ping ? &lds_w[wid][1][0] : &lds_w[wid][0][0];
    const float* l0 = lb + (size_t)row * 32;           // A row, mi=0
    const float* l1 = lb + (size_t)(row + 16) * 32;    // A row, mi=1

    // ---- A fragments from LDS: two 8-float runs per lane (ds_load_b128) --
    bf16x16 a[2];
#pragma unroll
    for (int mi = 0; mi < 2; ++mi) {
      const float* ab = (mi ? l1 : l0) + half8;
      const float4 f0 = *(const float4*)(ab);
      const float4 f1 = *(const float4*)(ab + 4);
      const float4 f2 = *(const float4*)(ab + 16);
      const float4 f3 = *(const float4*)(ab + 20);
      a[mi][0]  = f2bf(f0.x); a[mi][1]  = f2bf(f0.y);
      a[mi][2]  = f2bf(f0.z); a[mi][3]  = f2bf(f0.w);
      a[mi][4]  = f2bf(f1.x); a[mi][5]  = f2bf(f1.y);
      a[mi][6]  = f2bf(f1.z); a[mi][7]  = f2bf(f1.w);
      a[mi][8]  = f2bf(f2.x); a[mi][9]  = f2bf(f2.y);
      a[mi][10] = f2bf(f2.z); a[mi][11] = f2bf(f2.w);
      a[mi][12] = f2bf(f3.x); a[mi][13] = f2bf(f3.y);
      a[mi][14] = f2bf(f3.z); a[mi][15] = f2bf(f3.w);
    }

    // ---- B fragments (im2col gather, lane-coalesced) + 8 WMMAs ----------
#pragma unroll
    for (int j = 0; j < 4; ++j) {
      bf16x16 b;
#pragma unroll
      for (int i = 0; i < 16; ++i) {
        const int koff = (i < 8) ? (half8 + i) : (16 + half8 + (i - 8));
        const int k  = kc + koff;
        const int ci = k / (KS * KS);
        const int r9 = k - ci * (KS * KS);
        const int ky = r9 / KS;
        const int kx = r9 - ky * KS;
        const int iy = oy[j] * stride + ky - pad;
        const int ix = ox[j] * stride + kx - pad;
        float bv = 0.f;
        if (iy >= 0 && iy < inH && ix >= 0 && ix < inW)
          bv = in[(ci * inH + iy) * inW + ix];
        b[i] = f2bf(bv);
      }
      acc[0 * 4 + j] = __builtin_amdgcn_wmma_f32_16x16x32_bf16(
          false, a[0], false, b, (short)0, acc[0 * 4 + j], false, false);
      acc[1 * 4 + j] = __builtin_amdgcn_wmma_f32_16x16x32_bf16(
          false, a[1], false, b, (short)0, acc[1 * 4 + j], false, false);
    }
    ping ^= 1;
  }

  // ---- epilogue: optional residual add + ReLU, coalesced stores ----------
#pragma unroll
  for (int mi = 0; mi < 2; ++mi) {
#pragma unroll
    for (int j = 0; j < 4; ++j) {
      const int p = p0 + (j << 4) + row;
#pragma unroll
      for (int r = 0; r < 8; ++r) {
        const int co = co0 + (mi << 4) + r + half8;
        float v = acc[mi * 4 + j][r];
        if (res) v += res[co * HW + p];
        if (do_relu) v = v > 0.f ? v : 0.f;
        out[co * HW + p] = v;
      }
    }
  }
}

// ---------------------------------------------------------------------------
// Channel attention helpers
// ---------------------------------------------------------------------------
__global__ void channel_mean_k(const float* __restrict__ x, float* __restrict__ mean, int HW) {
  __shared__ float sm[256];
  const int c = blockIdx.x;
  float s = 0.f;
  for (int i = threadIdx.x; i < HW; i += 256) s += x[c * HW + i];
  sm[threadIdx.x] = s;
  __syncthreads();
  for (int o = 128; o > 0; o >>= 1) {
    if (threadIdx.x < o) sm[threadIdx.x] += sm[threadIdx.x + o];
    __syncthreads();
  }
  if (threadIdx.x == 0) mean[c] = sm[0] / (float)HW;
}

__global__ void att_sig_k(const float* __restrict__ mean, const float* __restrict__ att,
                          float* __restrict__ s, int C) {
  const int j = blockIdx.x * blockDim.x + threadIdx.x;
  if (j >= C) return;
  float a = 0.f;
  for (int c = 0; c < C; ++c) a += mean[c] * att[c * C + j];
  s[j] = 1.f / (1.f + expf(-a));
}

__global__ void scale_add_relu_k(const float* __restrict__ h, const float* __restrict__ s,
                                 const float* __restrict__ xin, float* __restrict__ xout,
                                 int tot, int HW) {
  const int i = blockIdx.x * blockDim.x + threadIdx.x;
  if (i >= tot) return;
  const int c = i / HW;
  float v = h[i] * s[c] + xin[i];
  xout[i] = v > 0.f ? v : 0.f;
}

// ---------------------------------------------------------------------------
// Elementwise utility kernels
// ---------------------------------------------------------------------------
__global__ void fill0_k(float* __restrict__ p, int n) {
  const int i = blockIdx.x * blockDim.x + threadIdx.x;
  if (i < n) p[i] = 0.f;
}
__global__ void copy_k(const float* __restrict__ a, float* __restrict__ b, int n) {
  const int i = blockIdx.x * blockDim.x + threadIdx.x;
  if (i < n) b[i] = a[i];
}
__global__ void add_k(const float* __restrict__ a, const float* __restrict__ b,
                      float* __restrict__ c, int n) {
  const int i = blockIdx.x * blockDim.x + threadIdx.x;
  if (i < n) c[i] = a[i] + b[i];
}

// ---------------------------------------------------------------------------
// Bilinear point sampling: feat [C,H,W] + coord [N,3] -> out [C,N]
// ---------------------------------------------------------------------------
__global__ void bilin_sample_k(const float* __restrict__ feat, const float* __restrict__ coord,
                               float* __restrict__ out, int C, int H, int W, int N,
                               float sh, float sw) {
  const int n = blockIdx.x * blockDim.x + threadIdx.x;
  if (n >= N) return;
  const float h = coord[n * 3 + 0] * sh;
  const float w = coord[n * 3 + 1] * sw;
  const float h0f = floorf(h), w0f = floorf(w);
  const float dh = h - h0f, dw = w - w0f;
  const int h0 = (int)h0f, w0 = (int)w0f;
  int   hi[4] = {h0, h0, h0 + 1, h0 + 1};
  int   wi[4] = {w0, w0 + 1, w0, w0 + 1};
  float wtv[4] = {(1.f - dh) * (1.f - dw), (1.f - dh) * dw,
                  dh * (1.f - dw),         dh * dw};
  int idx[4];
#pragma unroll
  for (int t = 0; t < 4; ++t) {
    const bool valid = (hi[t] >= 0) & (hi[t] < H) & (wi[t] >= 0) & (wi[t] < W);
    const int hc = min(max(hi[t], 0), H - 1);
    const int wc = min(max(wi[t], 0), W - 1);
    idx[t] = hc * W + wc;
    if (!valid) wtv[t] = 0.f;
  }
  const int HW = H * W;
  for (int c = 0; c < C; ++c) {
    const float* f = feat + c * HW;
    out[c * N + n] = wtv[0] * f[idx[0]] + wtv[1] * f[idx[1]] +
                     wtv[2] * f[idx[2]] + wtv[3] * f[idx[3]];
  }
}

// ---------------------------------------------------------------------------
// Voxel scatter-max: pfeat [C,N] (>=0, post-ReLU) into zero-init grid [C,H,W].
// Non-negative floats order identically as ints -> int atomicMax is exact.
// ---------------------------------------------------------------------------
__global__ void voxel_pool_k(const float* __restrict__ pf, const float* __restrict__ pind,
                             float* __restrict__ grid, int C, int N, int H, int W,
                             float sh, float sw) {
  const int n = blockIdx.x * blockDim.x + threadIdx.x;
  if (n >= N) return;
  const int h = (int)floorf(pind[n * 3 + 0] * sh);
  const int w = (int)floorf(pind[n * 3 + 1] * sw);
  if (h < 0 || h >= H || w < 0 || w >= W) return;
  const int idx = h * W + w;
  const int HW = H * W;
  for (int c = 0; c < C; ++c) {
    const float v = pf[c * N + n];
    atomicMax((int*)&grid[c * HW + idx], __float_as_int(v));
  }
}

// ---------------------------------------------------------------------------
// Align-corners bilinear upsample: x [C,inH,inW] -> out [C,outH,outW]
// ---------------------------------------------------------------------------
__global__ void upsample_ac_k(const float* __restrict__ x, float* __restrict__ out,
                              int C, int inH, int inW, int outH, int outW) {
  const int i = blockIdx.x * blockDim.x + threadIdx.x;
  const int tot = C * outH * outW;
  if (i >= tot) return;
  const int c = i / (outH * outW);
  const int r = i - c * (outH * outW);
  const int oy = r / outW, ox = r - oy * outW;
  const float sy = (outH > 1) ? (float)oy * (float)(inH - 1) / (float)(outH - 1) : 0.f;
  const float sx = (outW > 1) ? (float)ox * (float)(inW - 1) / (float)(outW - 1) : 0.f;
  const int y0 = (int)floorf(sy), x0 = (int)floorf(sx);
  const int y1 = min(y0 + 1, inH - 1), x1 = min(x0 + 1, inW - 1);
  const float fy = sy - (float)y0, fx = sx - (float)x0;
  const float* p = x + c * inH * inW;
  out[i] = (1.f - fy) * (1.f - fx) * p[y0 * inW + x0] +
           (1.f - fy) * fx * p[y0 * inW + x1] +
           fy * (1.f - fx) * p[y1 * inW + x0] +
           fy * fx * p[y1 * inW + x1];
}

// ---------------------------------------------------------------------------
// Small 1x1 conv with bias (aux heads, cout=3): scalar, bandwidth bound.
// ---------------------------------------------------------------------------
__global__ void aux1x1_k(const float* __restrict__ x, const float* __restrict__ w,
                         const float* __restrict__ b, float* __restrict__ out,
                         int cin, int cout, int HW) {
  const int i = blockIdx.x * blockDim.x + threadIdx.x;
  const int tot = cout * HW;
  if (i >= tot) return;
  const int co = i / HW;
  const int p  = i - co * HW;
  float a = b[co];
  for (int c = 0; c < cin; ++c) a += w[co * cin + c] * x[c * HW + p];
  out[i] = a;
}

// ---------------------------------------------------------------------------
// Host-side orchestration helpers
// ---------------------------------------------------------------------------
static inline int ceil_div(int a, int b) { return (a + b - 1) / b; }

static inline void conv(hipStream_t st, const float* in, const float* w,
                        const float* res, float* out, int cin, int cout,
                        int inH, int inW, int outH, int outW,
                        int ks, int stride, int pad, int relu) {
  const int tiles  = (cout / 32) * ((outH * outW) / 64);
  const int blocks = ceil_div(tiles, 8);   // 8 waves per 256-thread block
  if (ks == 3)
    conv_wmma_k<3><<<blocks, 256, 0, st>>>(in, w, res, out, cin, cout, inH, inW,
                                           outH, outW, stride, pad, relu);
  else
    conv_wmma_k<1><<<blocks, 256, 0, st>>>(in, w, res, out, cin, cout, inH, inW,
                                           outH, outW, stride, pad, relu);
}

static inline void basic_block(hipStream_t st, float* x, const float* w1,
                               const float* w2, const float* att, int C, int H,
                               int W, float* tA, float* tB, float* meanb,
                               float* sv) {
  conv(st, x, w1, nullptr, tA, C, C, H, W, H, W, 3, 1, 1, 1);   // h = relu(conv)
  if (!att) {
    conv(st, tA, w2, x, x, C, C, H, W, H, W, 3, 1, 1, 1);       // x = relu(conv+x)
  } else {
    conv(st, tA, w2, nullptr, tB, C, C, H, W, H, W, 3, 1, 1, 0);
    channel_mean_k<<<C, 256, 0, st>>>(tB, meanb, H * W);
    att_sig_k<<<ceil_div(C, 64), 64, 0, st>>>(meanb, att, sv, C);
    const int tot = C * H * W;
    scale_add_relu_k<<<ceil_div(tot, 256), 256, 0, st>>>(tB, sv, x, x, tot, H * W);
  }
}

// ---------------------------------------------------------------------------
// kernel_launch: inputs assumed flattened in setup_inputs() insertion order
// (descartes_feat_in, des_coord_t0, sph_coord_t0, temporal_res, then every
// params leaf in init_params() insertion order) -> 59 input pointers.
// ---------------------------------------------------------------------------
extern "C" void kernel_launch(void* const* d_in, const int* in_sizes, int n_in,
                              void* d_out, int out_size, void* d_ws, size_t ws_size,
                              hipStream_t stream) {
  const float* P[64];
  for (int i = 0; i < n_in && i < 64; ++i) P[i] = (const float*)d_in[i];

  const float* x_in      = P[0];               // [192,512,512]
  const float* des_coord = P[1];               // [N,3]
  const float* sph_coord = P[2];               // [N,3]
  const float* temporal  = P[3];               // [128,64,64]
  const int N = in_sizes[1] / 3;               // 131072

  // ---- workspace bump allocator (fp32 elements) ----
  float* ws = (float*)d_ws;
  size_t off = 0;
  auto alloc = [&](size_t n) { float* p = ws + off; off += n; return p; };
  float* des1  = alloc(32 * 65536);            // [32,256,256]
  float* blkA  = alloc(32 * 65536);            // block temp h
  float* blkB  = alloc(32 * 65536);            // block temp h2
  float* l1cat = alloc(64 * 65536);            // concat [des1 | sph1_as_des]
  float* l1f   = alloc(32 * 65536);            // l1_fused
  float* d1s   = alloc(32 * 32768);            // des1_as_sph [32,32,1024]
  float* sph1  = alloc(32 * 32768);            // persists to backprojection
  float* des2  = alloc(64 * 16384);            // [64,128,128]
  float* d2s   = alloc(64 * 8192);             // [64,16,512]
  float* sph2  = alloc(64 * 8192);
  float* l2cat = alloc(128 * 16384);
  float* l2f   = alloc(64 * 16384);
  float* des3a = alloc(128 * 4096);
  float* des3b = alloc(128 * 4096);
  float* des3  = alloc(128 * 4096);
  float* meanb = alloc(256);
  float* sv    = alloc(256);
  float* ptbuf = alloc(224 * 65536);           // point buffer; later reused as cat
  float* cat   = ptbuf;                        // decoder concat [224,256,256]
  float* t1    = alloc(128 * 65536);
  float* desout= alloc(64 * 65536);
  (void)ws_size; (void)out_size;

  hipStream_t st = stream;
  const int TB = 256;

  // ================= Level 1 (BEV 512 -> 256) =================
  conv(st, x_in, P[4], nullptr, des1, 192, 32, 512, 512, 256, 256, 3, 2, 1, 1);
  basic_block(st, des1, P[5], P[6],  nullptr, 32, 256, 256, blkA, blkB, meanb, sv);
  basic_block(st, des1, P[7], P[8],  nullptr, 32, 256, 256, blkA, blkB, meanb, sv);
  basic_block(st, des1, P[9], P[10], P[11],   32, 256, 256, blkA, blkB, meanb, sv);

  bilin_sample_k<<<ceil_div(N, TB), TB, 0, st>>>(des1, des_coord, ptbuf, 32, 256, 256, N, 0.5f, 0.5f);
  fill0_k<<<ceil_div(32 * 32768, TB), TB, 0, st>>>(d1s, 32 * 32768);
  voxel_pool_k<<<ceil_div(N, TB), TB, 0, st>>>(ptbuf, sph_coord, d1s, 32, N, 32, 1024, 0.5f, 0.5f);

  conv(st, d1s, P[12], nullptr, sph1, 32, 32, 32, 1024, 32, 1024, 3, 1, 1, 1);
  basic_block(st, sph1, P[13], P[14], nullptr, 32, 32, 1024, blkA, blkB, meanb, sv);
  basic_block(st, sph1, P[15], P[16], P[17],   32, 32, 1024, blkA, blkB, meanb, sv);

  bilin_sample_k<<<ceil_div(N, TB), TB, 0, st>>>(sph1, sph_coord, ptbuf, 32, 32, 1024, N, 0.5f, 0.5f);
  fill0_k<<<ceil_div(32 * 65536, TB), TB, 0, st>>>(l1cat + 32 * 65536, 32 * 65536);
  voxel_pool_k<<<ceil_div(N, TB), TB, 0, st>>>(ptbuf, des_coord, l1cat + 32 * 65536, 32, N, 256, 256, 0.5f, 0.5f);
  copy_k<<<ceil_div(32 * 65536, TB), TB, 0, st>>>(des1, l1cat, 32 * 65536);
  conv(st, l1cat, P[18], nullptr, l1f, 64, 32, 256, 256, 256, 256, 1, 1, 0, 1);

  // ================= Level 2 (256 -> 128) =================
  conv(st, l1f, P[19], nullptr, des2, 32, 64, 256, 256, 128, 128, 3, 2, 1, 1);
  basic_block(st, des2, P[20], P[21], nullptr, 64, 128, 128, blkA, blkB, meanb, sv);
  basic_block(st, des2, P[22], P[23], nullptr, 64, 128, 128, blkA, blkB, meanb, sv);
  basic_block(st, des2, P[24], P[25], nullptr, 64, 128, 128, blkA, blkB, meanb, sv);
  basic_block(st, des2, P[26], P[27], P[28],   64, 128, 128, blkA, blkB, meanb, sv);

  bilin_sample_k<<<ceil_div(N, TB), TB, 0, st>>>(des2, des_coord, ptbuf, 64, 128, 128, N, 0.25f, 0.25f);
  fill0_k<<<ceil_div(64 * 8192, TB), TB, 0, st>>>(d2s, 64 * 8192);
  voxel_pool_k<<<ceil_div(N, TB), TB, 0, st>>>(ptbuf, sph_coord, d2s, 64, N, 16, 512, 0.25f, 0.25f);

  conv(st, d2s, P[29], nullptr, sph2, 64, 64, 16, 512, 16, 512, 3, 1, 1, 1);
  basic_block(st, sph2, P[30], P[31], nullptr, 64, 16, 512, blkA, blkB, meanb, sv);
  basic_block(st, sph2, P[32], P[33], nullptr, 64, 16, 512, blkA, blkB, meanb, sv);
  basic_block(st, sph2, P[34], P[35], P[36],   64, 16, 512, blkA, blkB, meanb, sv);

  bilin_sample_k<<<ceil_div(N, TB), TB, 0, st>>>(sph2, sph_coord, ptbuf, 64, 16, 512, N, 0.25f, 0.25f);
  fill0_k<<<ceil_div(64 * 16384, TB), TB, 0, st>>>(l2cat + 64 * 16384, 64 * 16384);
  voxel_pool_k<<<ceil_div(N, TB), TB, 0, st>>>(ptbuf, des_coord, l2cat + 64 * 16384, 64, N, 128, 128, 0.25f, 0.25f);
  copy_k<<<ceil_div(64 * 16384, TB), TB, 0, st>>>(des2, l2cat, 64 * 16384);
  conv(st, l2cat, P[37], nullptr, l2f, 128, 64, 128, 128, 128, 128, 1, 1, 0, 1);

  // ================= Level 3 (128 -> 64) =================
  conv(st, l2f, P[38], nullptr, des3a, 64, 128, 128, 128, 64, 64, 3, 2, 1, 1);
  basic_block(st, des3a, P[39], P[40], nullptr, 128, 64, 64, blkA, blkB, meanb, sv);
  basic_block(st, des3a, P[41], P[42], nullptr, 128, 64, 64, blkA, blkB, meanb, sv);
  basic_block(st, des3a, P[43], P[44], nullptr, 128, 64, 64, blkA, blkB, meanb, sv);
  basic_block(st, des3a, P[45], P[46], nullptr, 128, 64, 64, blkA, blkB, meanb, sv);
  basic_block(st, des3a, P[47], P[48], P[49],   128, 64, 64, blkA, blkB, meanb, sv);
  add_k<<<ceil_div(128 * 4096, TB), TB, 0, st>>>(des3a, temporal, des3b, 128 * 4096);
  conv(st, des3b, P[50], nullptr, des3, 128, 128, 64, 64, 64, 64, 1, 1, 0, 1);

  // ================= Decoder (upsample into concat buffer) =================
  // NOTE: ptbuf is dead from here on; reuse it as `cat` [224,256,256].
  upsample_ac_k<<<ceil_div(32 * 65536, TB), TB, 0, st>>>(l1f,  cat,              32, 256, 256, 256, 256);
  upsample_ac_k<<<ceil_div(64 * 65536, TB), TB, 0, st>>>(l2f,  cat + 32 * 65536, 64, 128, 128, 256, 256);
  upsample_ac_k<<<ceil_div(128 * 65536, TB), TB, 0, st>>>(des3, cat + 96 * 65536, 128, 64, 64, 256, 256);

  conv(st, cat, P[51], nullptr, t1,     224, 128, 256, 256, 256, 256, 3, 1, 1, 1);
  conv(st, t1,  P[52], nullptr, desout, 128, 64,  256, 256, 256, 256, 3, 1, 1, 1);

  // ================= Outputs (concatenated in return order) =================
  float* dout = (float*)d_out;
  const size_t o0 = 0;                         // des_out_as_point [64,N]
  const size_t o1 = o0 + (size_t)64 * N;       // sph_out_as_point [32,N]
  const size_t o2 = o1 + (size_t)32 * N;       // aux1 [3,256,256]
  const size_t o3 = o2 + (size_t)3 * 65536;    // aux2
  const size_t o4 = o3 + (size_t)3 * 65536;    // aux3
  const size_t o5 = o4 + (size_t)3 * 65536;    // des3 [128,64,64]

  aux1x1_k<<<ceil_div(3 * 65536, TB), TB, 0, st>>>(cat,              P[53], P[54], dout + o2, 32, 3, 65536);
  aux1x1_k<<<ceil_div(3 * 65536, TB), TB, 0, st>>>(cat + 32 * 65536, P[55], P[56], dout + o3, 64, 3, 65536);
  aux1x1_k<<<ceil_div(3 * 65536, TB), TB, 0, st>>>(cat + 96 * 65536, P[57], P[58], dout + o4, 128, 3, 65536);

  bilin_sample_k<<<ceil_div(N, TB), TB, 0, st>>>(desout, des_coord, dout + o0, 64, 256, 256, N, 0.5f, 0.5f);
  bilin_sample_k<<<ceil_div(N, TB), TB, 0, st>>>(sph1,   sph_coord, dout + o1, 32, 32, 1024, N, 0.5f, 0.5f);
  copy_k<<<ceil_div(128 * 4096, TB), TB, 0, st>>>(des3, dout + o5, 128 * 4096);
}
